// LinearAttentionBlock_65481071410945
// MI455X (gfx1250) — matrix operand order repaired and checked by reading, not compiled, not run on hardware
//
#include <hip/hip_runtime.h>
#include <hip/hip_bf16.h>

typedef __attribute__((ext_vector_type(16))) _Float16 v16h;
typedef __attribute__((ext_vector_type(8)))  _Float16 v8h;
typedef __attribute__((ext_vector_type(8)))  float    v8f;
typedef __attribute__((ext_vector_type(4)))  int      v4i;

namespace {
constexpr int B  = 4, L = 4096, D = 2048, H = 16, HD = 128;
constexpr int BH = B * H;               // 64
constexpr int NKV = BH * HD * HD;       // 1,048,576 f32 elems per full KV set
constexpr int NKS = BH * HD;            // 8192
constexpr int KV_CHUNKS = 4;
constexpr int LCHUNK = L / KV_CHUNKS;   // 1024
}

#define GLOBAL_AS __attribute__((address_space(1)))
#define LDS_AS    __attribute__((address_space(3)))

// ---------- async global->LDS copy (gfx1250 GLOBAL_LOAD_ASYNC_TO_LDS_B128) ----------
#if defined(__gfx1250__) && __has_builtin(__builtin_amdgcn_global_load_async_to_lds_b128)
#define HAVE_ASYNC_LDS 1
__device__ static inline void async_copy_b128(const _Float16* g, _Float16* l) {
  __builtin_amdgcn_global_load_async_to_lds_b128(
      (GLOBAL_AS v4i*)(g), (LDS_AS v4i*)(l), /*offset=*/0, /*cpol=*/0);
}
__device__ static inline void async_wait0() {
#if __has_builtin(__builtin_amdgcn_s_wait_asynccnt)
  __builtin_amdgcn_s_wait_asynccnt(0);
#else
  asm volatile("s_wait_asynccnt 0x0" ::: "memory");
#endif
}
#else
#define HAVE_ASYNC_LDS 0
__device__ static inline void async_copy_b128(const _Float16* g, _Float16* l) {
  *(v8h*)l = *(const v8h*)g;            // synchronous fallback
}
__device__ static inline void async_wait0() {}
#endif

// ---------- WMMA helpers (gfx1250 wave32, 16x16x32 f16 -> f32) ----------

__device__ static inline v8f wmma32(v16h a, v16h b, v8f c) {
  return __builtin_amdgcn_wmma_f32_16x16x32_f16(
      /*neg_a=*/false, a, /*neg_b=*/false, b,
      /*c_mod=*/(short)0, c, /*reuse_a=*/false, /*reuse_b=*/false);
}

// A fragment (16x32, row-major [m][k] in LDS, ld in halfs, ld even).
__device__ static inline v16h frag_a_mk(const _Float16* s, int ld, int mBase) {
  const int lane = threadIdx.x & 31;
  const int m  = mBase + (lane & 15);
  const int hi = lane >> 4;
  v16h a;
  const unsigned int* row = (const unsigned int*)(s + m * ld);
#pragma unroll
  for (int v = 0; v < 8; ++v) {
    // element pair K = v*2 + hi*16  (always even)
    const int kpair = v + hi * 8;
    ((unsigned int*)&a)[v] = row[kpair];
  }
  return a;
}

// B fragment (32x16) from LDS stored [n][k] (contiguous k per n-row).
__device__ static inline v16h frag_b_nk(const _Float16* s, int ld, int nBase) {
  const int lane = threadIdx.x & 31;
  const int n  = nBase + (lane & 15);
  const int hi = lane >> 4;
  v16h b;
  const unsigned int* row = (const unsigned int*)(s + n * ld);
#pragma unroll
  for (int v = 0; v < 8; ++v)
    ((unsigned int*)&b)[v] = row[v + hi * 8];
  return b;
}

// B fragment (32x16) from LDS stored [k][n] (row-major k, contiguous n).
__device__ static inline v16h frag_b_kn(const _Float16* s, int ld, int nBase) {
  const int lane = threadIdx.x & 31;
  const int n  = nBase + (lane & 15);
  const int hi = lane >> 4;
  v16h b;
#pragma unroll
  for (int v = 0; v < 8; ++v) {
    const int k = v * 2 + hi * 16;
    b[v * 2]     = s[k * ld + n];
    b[v * 2 + 1] = s[(k + 1) * ld + n];
  }
  return b;
}

// ---------------- Kernel 1: fused QKV projection + elu+1 ----------------
__global__ __launch_bounds__(256) void k_qkv(
    const float* __restrict__ x,
    const float* __restrict__ Wq, const float* __restrict__ bq,
    const float* __restrict__ Wk, const float* __restrict__ bk,
    const float* __restrict__ Wv, const float* __restrict__ bv,
    _Float16* __restrict__ Qh, _Float16* __restrict__ Kh, _Float16* __restrict__ Vh)
{
  constexpr int AS = 40, BS = 40;          // half strides; 80B = 5*16 keeps b128 aligned
  __shared__ _Float16 As[64 * AS];         // [m 64][k 32]
  __shared__ _Float16 Bs[128 * BS];        // [n 128][k 32]

  const int which = blockIdx.z;
  const float* __restrict__ W    = (which == 0) ? Wq : (which == 1) ? Wk : Wv;
  const float* __restrict__ bias = (which == 0) ? bq : (which == 1) ? bk : bv;
  _Float16* __restrict__ Out     = (which == 0) ? Qh : (which == 1) ? Kh : Vh;

  const int mBase = blockIdx.y * 64;
  const int nBase = blockIdx.x * 128;
  const int tid  = threadIdx.x;
  const int lane = tid & 31, hi = lane >> 4, nn = lane & 15;
  const int wave = tid >> 5;
  const int wm = (wave & 1) * 32;
  const int wn = (wave >> 1) * 32;

  v8f c00 = {}, c01 = {}, c10 = {}, c11 = {};

  const int ar = tid >> 2, ak = (tid & 3) * 8;   // A stage: 64 rows x 32 k
  const int br = tid >> 1, bkk = (tid & 1) * 16; // B stage: 128 rows x 32 k

  for (int k0 = 0; k0 < D; k0 += 32) {
    __syncthreads();
    {
      const float4* p = (const float4*)(x + (size_t)(mBase + ar) * D + k0 + ak);
      const float4 f0 = p[0], f1 = p[1];
      v8h h;
      h[0]=(_Float16)f0.x; h[1]=(_Float16)f0.y; h[2]=(_Float16)f0.z; h[3]=(_Float16)f0.w;
      h[4]=(_Float16)f1.x; h[5]=(_Float16)f1.y; h[6]=(_Float16)f1.z; h[7]=(_Float16)f1.w;
      *(v8h*)&As[ar * AS + ak] = h;
    }
    {
      const float4* p = (const float4*)(W + (size_t)(nBase + br) * D + k0 + bkk);
      const float4 f0 = p[0], f1 = p[1], f2 = p[2], f3 = p[3];
      v8h h0, h1;
      h0[0]=(_Float16)f0.x; h0[1]=(_Float16)f0.y; h0[2]=(_Float16)f0.z; h0[3]=(_Float16)f0.w;
      h0[4]=(_Float16)f1.x; h0[5]=(_Float16)f1.y; h0[6]=(_Float16)f1.z; h0[7]=(_Float16)f1.w;
      h1[0]=(_Float16)f2.x; h1[1]=(_Float16)f2.y; h1[2]=(_Float16)f2.z; h1[3]=(_Float16)f2.w;
      h1[4]=(_Float16)f3.x; h1[5]=(_Float16)f3.y; h1[6]=(_Float16)f3.z; h1[7]=(_Float16)f3.w;
      *(v8h*)&Bs[br * BS + bkk]     = h0;
      *(v8h*)&Bs[br * BS + bkk + 8] = h1;
    }
    __syncthreads();
    const v16h a0 = frag_a_mk(As, AS, wm);
    const v16h a1 = frag_a_mk(As, AS, wm + 16);
    const v16h b0 = frag_b_nk(Bs, BS, wn);
    const v16h b1 = frag_b_nk(Bs, BS, wn + 16);
    c00 = wmma32(a0, b0, c00);
    c01 = wmma32(a0, b1, c01);
    c10 = wmma32(a1, b0, c10);
    c11 = wmma32(a1, b1, c11);
  }

  auto epi = [&](const v8f& c, int mo, int no) {
    const int col = nBase + wn + no + nn;
    const float bv_ = bias[col];
    const int hh = col >> 7, dd = col & (HD - 1);
#pragma unroll
    for (int v = 0; v < 8; ++v) {
      const int gm = mBase + wm + mo + v + hi * 8;   // global row in [0, B*L)
      const int bb = gm >> 12;                       // / L
      const int ll = gm & (L - 1);
      float val = c[v] + bv_;
      if (which < 2) val = (val > 0.f) ? (val + 1.f) : __expf(val);  // elu+1
      Out[(((size_t)bb * H + hh) * L + ll) * HD + dd] = (_Float16)val;
    }
  };
  epi(c00, 0, 0); epi(c01, 0, 16); epi(c10, 16, 0); epi(c11, 16, 16);
}

// ---------- Kernel 2: KV = K^T V partials (split-K over L) + Ksum partials ----------
__global__ __launch_bounds__(256) void k_kv(
    const _Float16* __restrict__ Kh, const _Float16* __restrict__ Vh,
    float* __restrict__ KVpart, float* __restrict__ KSpart)
{
  constexpr int AS = 40;
  __shared__ _Float16 As[128 * AS];   // K^T tile: [d 128][l 32]
  __shared__ _Float16 Bs[32 * 128];   // V tile:   [l 32][e 128]

  const int bh = blockIdx.x;
  const int chunk = blockIdx.y;
  const int tid = threadIdx.x;
  const int lane = tid & 31, hi = lane >> 4, nn = lane & 15;
  const int wave = tid >> 5;
  const int wm = (wave & 1) * 64;     // 2 waves over d
  const int wn = (wave >> 1) * 32;    // 4 waves over e

  v8f acc[4][2];
  const v8f zz = {};
#pragma unroll
  for (int i = 0; i < 4; ++i) { acc[i][0] = zz; acc[i][1] = zz; }

  const int lr = tid >> 3;            // 32 l rows
  const int dc = (tid & 7) * 16;      // 16 d per thread
  const int lBase = chunk * LCHUNK;

  for (int l0 = 0; l0 < LCHUNK; l0 += 32) {
    __syncthreads();
    {
      const size_t src = ((size_t)bh * L + lBase + l0 + lr) * HD + dc;
      // V tile: pure f16 copy -> async LDS path (no VGPR round-trip)
      async_copy_b128(&Vh[src],     &Bs[lr * 128 + dc]);
      async_copy_b128(&Vh[src + 8], &Bs[lr * 128 + dc + 8]);
      // K tile needs a transpose: classic VGPR path
      const v8h ka = *(const v8h*)&Kh[src];
      const v8h kb = *(const v8h*)&Kh[src + 8];
#pragma unroll
      for (int j = 0; j < 8; ++j) {             // transpose K into [d][l]
        As[(dc + j)     * AS + lr] = ka[j];
        As[(dc + 8 + j) * AS + lr] = kb[j];
      }
    }
    async_wait0();
    __syncthreads();
    const v16h b0 = frag_b_kn(Bs, 128, wn);
    const v16h b1 = frag_b_kn(Bs, 128, wn + 16);
#pragma unroll
    for (int i = 0; i < 4; ++i) {
      const v16h a = frag_a_mk(As, AS, wm + i * 16);
      acc[i][0] = wmma32(a, b0, acc[i][0]);
      acc[i][1] = wmma32(a, b1, acc[i][1]);
    }
  }

  float* dst = KVpart + (size_t)chunk * NKV + (size_t)bh * HD * HD;
#pragma unroll
  for (int i = 0; i < 4; ++i)
#pragma unroll
    for (int j = 0; j < 2; ++j)
#pragma unroll
      for (int v = 0; v < 8; ++v) {
        const int m = wm + i * 16 + v + hi * 8;
        const int e = wn + j * 16 + nn;
        dst[m * HD + e] = acc[i][j][v];
      }

  if (tid < HD) {                       // Ksum partial for this chunk
    float s = 0.f;
    for (int l = 0; l < LCHUNK; ++l)
      s += (float)Kh[((size_t)bh * L + lBase + l) * HD + tid];
    KSpart[(size_t)chunk * NKS + bh * HD + tid] = s;
  }
}

// ---------------- Kernel 3: reduce split-K partials ----------------
__global__ __launch_bounds__(256) void k_reduce(
    const float* __restrict__ KVpart, const float* __restrict__ KSpart,
    _Float16* __restrict__ KVf16, float* __restrict__ Ksum)
{
  const int i = blockIdx.x * blockDim.x + threadIdx.x;
  if (i < NKV) {
    float s = 0.f;
#pragma unroll
    for (int c = 0; c < KV_CHUNKS; ++c) s += KVpart[(size_t)c * NKV + i];
    KVf16[i] = (_Float16)s;
  }
  if (i < NKS) {
    float s = 0.f;
#pragma unroll
    for (int c = 0; c < KV_CHUNKS; ++c) s += KSpart[(size_t)c * NKS + i];
    Ksum[i] = s;
  }
}

// ---------- Kernel 4: out = (Q @ KV) / (Q . Ksum + 1e-6), scatter to (b,l,D) ----------
__global__ __launch_bounds__(256) void k_attn(
    const _Float16* __restrict__ Qh, const _Float16* __restrict__ KVf16,
    const float* __restrict__ Ksum, _Float16* __restrict__ attn)
{
  constexpr int AS = 40;
  __shared__ _Float16 As[64 * AS];     // Q tile [l 64][d 32]
  __shared__ _Float16 Bs[128 * 128];   // full KV [d 128][e 128]
  __shared__ float ksum_s[HD];
  __shared__ float denom_s[64];

  const int ltile = blockIdx.x;
  const int bh = blockIdx.y;
  const int tid = threadIdx.x;
  const int lane = tid & 31, hi = lane >> 4, nn = lane & 15;
  const int wave = tid >> 5;
  const int wm = (wave & 1) * 32;
  const int wn = (wave >> 1) * 32;

  {
    const _Float16* src = KVf16 + (size_t)bh * HD * HD;
    for (int i = tid * 8; i < HD * HD; i += 256 * 8)
      async_copy_b128(&src[i], &Bs[i]);          // 32 KB f16 tile via async LDS
    if (tid < HD) ksum_s[tid] = Ksum[bh * HD + tid];
  }
  async_wait0();
  __syncthreads();

  if (tid < 64) {
    const int l = ltile * 64 + tid;
    const _Float16* q = Qh + ((size_t)bh * L + l) * HD;
    float s = 0.f;
    for (int d = 0; d < HD; ++d) s += (float)q[d] * ksum_s[d];
    denom_s[tid] = s + 1e-6f;
  }

  v8f c00 = {}, c01 = {}, c10 = {}, c11 = {};
  const int ar = tid >> 2, ak = (tid & 3) * 8;

  for (int k0 = 0; k0 < HD; k0 += 32) {
    __syncthreads();
    async_copy_b128(&Qh[((size_t)bh * L + ltile * 64 + ar) * HD + k0 + ak],
                    &As[ar * AS + ak]);
    async_wait0();
    __syncthreads();
    const v16h a0 = frag_a_mk(As, AS, wm);
    const v16h a1 = frag_a_mk(As, AS, wm + 16);
    const v16h b0 = frag_b_kn(Bs + k0 * 128, 128, wn);
    const v16h b1 = frag_b_kn(Bs + k0 * 128, 128, wn + 16);
    c00 = wmma32(a0, b0, c00);
    c01 = wmma32(a0, b1, c01);
    c10 = wmma32(a1, b0, c10);
    c11 = wmma32(a1, b1, c11);
  }

  const int b = bh >> 4, h = bh & (H - 1);
  auto epi = [&](const v8f& c, int mo, int no) {
    const int e = wn + no + nn;
#pragma unroll
    for (int v = 0; v < 8; ++v) {
      const int lm = wm + mo + v + hi * 8;
      const int l  = ltile * 64 + lm;
      const float val = c[v] / denom_s[lm];
      attn[((size_t)b * L + l) * D + h * HD + e] = (_Float16)val;
    }
  };
  epi(c00, 0, 0); epi(c01, 0, 16); epi(c10, 16, 0); epi(c11, 16, 16);
}

// ---------------- Kernel 5: y = attn @ Wo^T + bo (fp32 out) ----------------
__global__ __launch_bounds__(256) void k_out(
    const _Float16* __restrict__ attn, const float* __restrict__ Wo,
    const float* __restrict__ bo, float* __restrict__ out)
{
  constexpr int AS = 40, BS = 40;
  __shared__ _Float16 As[64 * AS];
  __shared__ _Float16 Bs[128 * BS];

  const int mBase = blockIdx.y * 64;
  const int nBase = blockIdx.x * 128;
  const int tid = threadIdx.x;
  const int lane = tid & 31, hi = lane >> 4, nn = lane & 15;
  const int wave = tid >> 5;
  const int wm = (wave & 1) * 32;
  const int wn = (wave >> 1) * 32;

  v8f c00 = {}, c01 = {}, c10 = {}, c11 = {};
  const int ar = tid >> 2, ak = (tid & 3) * 8;
  const int br = tid >> 1, bkk = (tid & 1) * 16;

  for (int k0 = 0; k0 < D; k0 += 32) {
    __syncthreads();
    // attn tile is already f16: async LDS copy
    async_copy_b128(&attn[(size_t)(mBase + ar) * D + k0 + ak], &As[ar * AS + ak]);
    {
      const float4* p = (const float4*)(Wo + (size_t)(nBase + br) * D + k0 + bkk);
      const float4 f0 = p[0], f1 = p[1], f2 = p[2], f3 = p[3];
      v8h h0, h1;
      h0[0]=(_Float16)f0.x; h0[1]=(_Float16)f0.y; h0[2]=(_Float16)f0.z; h0[3]=(_Float16)f0.w;
      h0[4]=(_Float16)f1.x; h0[5]=(_Float16)f1.y; h0[6]=(_Float16)f1.z; h0[7]=(_Float16)f1.w;
      h1[0]=(_Float16)f2.x; h1[1]=(_Float16)f2.y; h1[2]=(_Float16)f2.z; h1[3]=(_Float16)f2.w;
      h1[4]=(_Float16)f3.x; h1[5]=(_Float16)f3.y; h1[6]=(_Float16)f3.z; h1[7]=(_Float16)f3.w;
      *(v8h*)&Bs[br * BS + bkk]     = h0;
      *(v8h*)&Bs[br * BS + bkk + 8] = h1;
    }
    async_wait0();
    __syncthreads();
    const v16h a0 = frag_a_mk(As, AS, wm);
    const v16h a1 = frag_a_mk(As, AS, wm + 16);
    const v16h b0 = frag_b_nk(Bs, BS, wn);
    const v16h b1 = frag_b_nk(Bs, BS, wn + 16);
    c00 = wmma32(a0, b0, c00);
    c01 = wmma32(a0, b1, c01);
    c10 = wmma32(a1, b0, c10);
    c11 = wmma32(a1, b1, c11);
  }

  auto epi = [&](const v8f& c, int mo, int no) {
    const int col = nBase + wn + no + nn;
    const float bb = bo[col];
#pragma unroll
    for (int v = 0; v < 8; ++v) {
      const int gm = mBase + wm + mo + v + hi * 8;
      out[(size_t)gm * D + col] = c[v] + bb;
    }
  };
  epi(c00, 0, 0); epi(c01, 0, 16); epi(c10, 16, 0); epi(c11, 16, 16);
}

// ------------------------------- launch -------------------------------
extern "C" void kernel_launch(void* const* d_in, const int* in_sizes, int n_in,
                              void* d_out, int out_size, void* d_ws, size_t ws_size,
                              hipStream_t stream) {
  (void)in_sizes; (void)n_in; (void)out_size; (void)ws_size;
  const float* x  = (const float*)d_in[0];
  const float* Wq = (const float*)d_in[1];
  const float* bq = (const float*)d_in[2];
  const float* Wk = (const float*)d_in[3];
  const float* bk = (const float*)d_in[4];
  const float* Wv = (const float*)d_in[5];
  const float* bv = (const float*)d_in[6];
  const float* Wo = (const float*)d_in[7];
  const float* bo = (const float*)d_in[8];
  float* out = (float*)d_out;

  const size_t SZ = (size_t)B * L * D;        // f16 elements per activation tensor
  _Float16* Qh    = (_Float16*)d_ws;
  _Float16* Kh    = Qh + SZ;
  _Float16* Vh    = Kh + SZ;
  _Float16* KVf16 = Vh + SZ;                  // NKV halfs
  float* KVpart = (float*)(KVf16 + (size_t)NKV);
  float* KSpart = KVpart + (size_t)KV_CHUNKS * NKV;
  float* Ksum   = KSpart + (size_t)KV_CHUNKS * NKS;
  _Float16* attn = Kh;                        // Kh is dead after k_kv; reuse for attn

  k_qkv<<<dim3(D / 128, (B * L) / 64, 3), 256, 0, stream>>>(
      x, Wq, bq, Wk, bk, Wv, bv, Qh, Kh, Vh);
  k_kv<<<dim3(BH, KV_CHUNKS), 256, 0, stream>>>(Kh, Vh, KVpart, KSpart);
  k_reduce<<<(NKV + 255) / 256, 256, 0, stream>>>(KVpart, KSpart, KVf16, Ksum);
  k_attn<<<dim3(L / 64, BH), 256, 0, stream>>>(Qh, KVf16, Ksum, attn);
  k_out<<<dim3(D / 128, (B * L) / 64), 256, 0, stream>>>(attn, Wo, bo, out);
}
